// FNO3D_27977416966518
// MI455X (gfx1250) — matrix operand order repaired and checked by reading, not compiled, not run on hardware
//
#include <hip/hip_runtime.h>

typedef __bf16 bf16;
typedef __attribute__((ext_vector_type(16))) __bf16 v16bf;
typedef __attribute__((ext_vector_type(8)))  float  v8f;
typedef __attribute__((ext_vector_type(4)))  unsigned int u32x4;
typedef __attribute__((ext_vector_type(8)))  int i32x8;
typedef __attribute__((ext_vector_type(4)))  int i32x4;

#define SP 262144L /* 64*64*64 spatial points */

// ---------- workspace layout (in floats) ----------
static constexpr long OFF_BW   = 0;        // 64x16 forward rDFT (W axis)
static constexpr long OFF_TREH = 1024;     // 16x64 fwd DFT re (H axis)
static constexpr long OFF_TIMH = 2048;     // 16x64 fwd DFT im
static constexpr long OFF_TRED = 3072;     // 16x64 fwd DFT re (D axis)
static constexpr long OFF_TIMD = 4096;     // 16x64 fwd DFT im
static constexpr long OFF_EDRE = 5120;     // 64x32 inv DFT re (D), cols>=16 zero
static constexpr long OFF_EDIM = 7168;
static constexpr long OFF_EHRE = 9216;     // 64x32 inv DFT re (H)
static constexpr long OFF_EHIM = 11264;
static constexpr long OFF_MW   = 13312;    // 32x64 inv rDFT (W), rows>=16 zero
static constexpr long OFF_SUM  = 15360;
static constexpr long OFF_SUMSQ= 15392;
static constexpr long OFF_MU   = 15424;
static constexpr long OFF_RSIG = 15456;
static constexpr long OFF_H0   = 16384;                  // h ping   (33.5M f)
static constexpr long OFF_HN   = OFF_H0 + 33554432;      // h pong
static constexpr long OFF_G1   = OFF_HN + 33554432;      // [b,c,d,h,16]
static constexpr long OFF_G2   = OFF_G1 + 8388608;       // [b,c,d,ky,16]
static constexpr long OFF_G3   = OFF_G2 + 2097152;       // [b,c,kz,ky,16]
static constexpr long OFF_S    = OFF_G3 + 524288;        // [b,o,kz,ky,16]
static constexpr long OFF_HH1  = OFF_S  + 524288;        // [b,o,d,ky,16]
static constexpr long OFF_HH2  = OFF_HH1 + 2097152;      // [b,o,d,h,16]

// fast fp32->bf16 (truncation: one half-move instead of bfe+add3+mov RNE chain)
__device__ __forceinline__ bf16 f2bf(float f) {
  union { float f; unsigned u; } v; v.f = f;
  union { unsigned short s; bf16 b; } o; o.s = (unsigned short)(v.u >> 16);
  return o.b;
}

__device__ __forceinline__ v8f wmma_bf16(v16bf a, v16bf b, v8f c) {
  return __builtin_amdgcn_wmma_f32_16x16x32_bf16(false, a, false, b, (short)0, c,
                                                 false, false);
}

// ---------- Tensor Data Mover: async 2-D fp32 tile, global -> LDS ----------
// D# per CDNA5 ISA 8.3-8.6: group0 = {count=1|..., lds_addr, ga_lo,
// ga[56:32]|type=2<<30}; group1 packs data_size=4B, tensor/tile dims, strides.
// tensor_dim == tile_dim so nothing is OOB; row stride carried in
// tensor_dim0_stride (may exceed tile width, e.g. channel-strided panels).
// This toolchain exposes the 6-arg builtin (extra int32x8 descriptor group).
__device__ __forceinline__ void tdm_load_2d(void* lds, const void* glb,
                                            unsigned tx, unsigned ty,
                                            unsigned long long stride_elts) {
  unsigned long long ga = (unsigned long long)glb;
  unsigned lo = (unsigned)(__SIZE_TYPE__)lds;   // flat LDS addr: low 32 = offset
  u32x4 g0 = { 1u, lo, (unsigned)ga,
               ((unsigned)((ga >> 32) & 0x1FFFFFFull)) | 0x80000000u };
  i32x8 g1 = { (int)0x00020000u,                       // data_size = 4 bytes
               (int)((tx & 0xFFFFu) << 16),            // tensor_dim0[15:0]
               (int)((ty & 0xFFFFu) << 16),            // tensor_dim1[15:0]
               (int)((tx & 0xFFFFu) << 16),            // tile_dim0
               (int)(ty & 0xFFFFu),                    // tile_dim1, tile_dim2=0
               (int)(unsigned)(stride_elts & 0xFFFFFFFFull),
               (int)(unsigned)((stride_elts >> 32) & 0xFFFFull),
               0 };
  i32x4 z4 = { 0, 0, 0, 0 };
  i32x8 z8 = { 0, 0, 0, 0, 0, 0, 0, 0 };
  __builtin_amdgcn_tensor_load_to_lds(g0, g1, z4, z4, z8, 0);
}

// A fragment (16x32 bf16, M x K) from fp32 LDS, row-major leading-dim lda.
// ISA layout: lanes 0-15 rows M=lane, K={k0..k0+7, k0+16..k0+23};
//             lanes 16-31 rows M=lane-16, K={k0+8..k0+15, k0+24..k0+31}.
__device__ __forceinline__ v16bf load_a_ldsf(const float* A, int lda, int m0, int k0,
                                             int lane) {
  const float* p = A + (m0 + (lane & 15)) * lda;
  int kb = k0 + ((lane & 16) ? 8 : 0);
  v16bf r;
#pragma unroll
  for (int v = 0; v < 8; ++v) {
    int kp = kb + 2 * (v & 3) + ((v >> 2) << 4);
    r[2 * v]     = f2bf(p[kp]);
    r[2 * v + 1] = f2bf(p[kp + 1]);
  }
  return r;
}

// A fragment from a contiguous global fp32 row; elements with k>=kvalid are 0.
__device__ __forceinline__ v16bf load_a_row(const float* arow, int k0, int kvalid,
                                            int lane) {
  int kb = k0 + ((lane & 16) ? 8 : 0);
  v16bf r;
#pragma unroll
  for (int v = 0; v < 8; ++v) {
    int kp = kb + 2 * (v & 3) + ((v >> 2) << 4);
    r[2 * v]     = (kp     < kvalid) ? f2bf(arow[kp])     : f2bf(0.f);
    r[2 * v + 1] = (kp + 1 < kvalid) ? f2bf(arow[kp + 1]) : f2bf(0.f);
  }
  return r;
}

// B fragment (32x16 bf16, K x N) from fp32 LDS. lanes 0-15: K={0-7,16-23} (col=lane),
// lanes 16-31: K={8-15,24-31}; VGPR v holds K pair kb+2(v&3)+16(v>>2), +1.
__device__ __forceinline__ v16bf load_b_ldsf(const float* B, int ldb, int k0, int n0,
                                             int lane) {
  int col = n0 + (lane & 15);
  int kb  = k0 + ((lane & 16) ? 8 : 0);
  v16bf r;
#pragma unroll
  for (int v = 0; v < 8; ++v) {
    int k = kb + 2 * (v & 3) + ((v >> 2) << 4);
    r[2 * v]     = f2bf(B[k * ldb + col]);
    r[2 * v + 1] = f2bf(B[(k + 1) * ldb + col]);
  }
  return r;
}

// B fragment from global fp32, row stride ldb, rows clamped to kmax (clamped rows
// are annihilated by zero A columns). swap: column pairs become (-im, re) for the
// complex-multiply second term.
__device__ __forceinline__ v16bf load_b_glb(const float* B, long ldb, int k0, int kmax,
                                            int j, bool swap, int lane) {
  int col   = swap ? (j ^ 1) : j;
  float sgn = (swap && !(j & 1)) ? -1.f : 1.f;
  int kb = k0 + ((lane & 16) ? 8 : 0);
  v16bf r;
#pragma unroll
  for (int v = 0; v < 8; ++v) {
    int ka = kb + 2 * (v & 3) + ((v >> 2) << 4);
    int kc = ka > kmax ? kmax : ka;
    int kd = (ka + 1) > kmax ? kmax : (ka + 1);
    r[2 * v]     = f2bf(sgn * B[(long)kc * ldb + col]);
    r[2 * v + 1] = f2bf(sgn * B[(long)kd * ldb + col]);
  }
  return r;
}

// ---------- twiddle / weight matrices ----------
__global__ void build_mats_k(float* __restrict__ W) {
  int i = blockIdx.x * blockDim.x + threadIdx.x;
  const float P2 = 6.2831853071795864f;
  if (i < 1024) {
    { // forward rDFT along W axis: 64 x 16, col j=2k:cos, 2k+1:-sin
      int w = i >> 4, j = i & 15, k = j >> 1;
      float th = P2 * (float)(k * w) / 64.f;
      W[OFF_BW + i] = (j & 1) ? -sinf(th) : cosf(th);
    }
    { // forward DFT matrices (H and D axes, identical): 16 x 64
      int ky = i >> 6, h = i & 63;
      float th = P2 * (float)(ky * h) / 64.f;
      float c = cosf(th), s = -sinf(th);
      W[OFF_TREH + i] = c; W[OFF_TIMH + i] = s;
      W[OFF_TRED + i] = c; W[OFF_TIMD + i] = s;
    }
  }
  if (i < 2048) {
    { // inverse DFT (D and H axes): 64 x 32, cols>=16 zero, scale 1/64
      int d = i >> 5, kz = i & 31;
      float th = P2 * (float)(kz * d) / 64.f;
      float sc = (kz < 16) ? (1.f / 64.f) : 0.f;
      W[OFF_EDRE + i] = sc * cosf(th); W[OFF_EDIM + i] = sc * sinf(th);
      W[OFF_EHRE + i] = sc * cosf(th); W[OFF_EHIM + i] = sc * sinf(th);
    }
    { // inverse rDFT along W (Hermitian fold): 32 x 64, rows>=16 zero
      int j = i >> 6, w = i & 63, k = j >> 1;
      float a = (j >= 16) ? 0.f : ((k == 0) ? (1.f / 64.f) : (2.f / 64.f));
      float th = P2 * (float)(k * w) / 64.f;
      W[OFF_MW + i] = (j & 1) ? (-a * sinf(th)) : (a * cosf(th));
    }
  }
}

// ---------- conv1x1 as WMMA GEMM: out[b,o,s] = sum_c Wm[o,c] in[b,c,s] + bias ----
// TDMP: weights AND the (channel-strided) 64-wide activation panel arrive in LDS
// via tensor_load_to_lds (requires C_in == KPAD, i.e. no K zero-padding).
template <int KPAD, bool TDMP>
__global__ void conv_wmma_k(const float* __restrict__ Wm, const float* __restrict__ bias,
                            const float* __restrict__ in, const float* __restrict__ resid,
                            float* __restrict__ out, int C_out, int C_in) {
  __shared__ float AL[64 * KPAD];
  __shared__ float BL[TDMP ? 64 * 64 : 1];

  long chunk = blockIdx.x;
  long b = chunk >> 12;                   // 4096 chunks of 64 cols per batch
  long col0 = (chunk & 4095) * 64;
  const float* Bglb = in + b * C_in * SP + col0;

  if constexpr (TDMP) {
    // zero only rows >= C_out (disjoint from the TDM-written region)
    for (int i = threadIdx.x + C_out * KPAD; i < 64 * KPAD; i += blockDim.x)
      AL[i] = 0.f;
    if (threadIdx.x < 32) {
      tdm_load_2d(AL, Wm, KPAD, (unsigned)C_out, KPAD);            // weight tile
      tdm_load_2d(BL, Bglb, 64, (unsigned)C_in, (unsigned long long)SP); // act panel
      __builtin_amdgcn_s_wait_tensorcnt(0);
    }
  } else {
    for (int i = threadIdx.x; i < 64 * KPAD; i += blockDim.x) {
      int row = i / KPAD, k = i - row * KPAD;
      AL[i] = (row < C_out && k < C_in) ? Wm[row * C_in + k] : 0.f;
    }
  }
  __syncthreads();

  int wave = threadIdx.x >> 5, lane = threadIdx.x & 31, j = lane & 15;
  int mtiles = (C_out + 15) >> 4;
  for (int mt = 0; mt < mtiles; ++mt) {
    v8f acc = {};
#pragma unroll
    for (int ks = 0; ks < KPAD; ks += 32) {
      v16bf a = load_a_ldsf(AL, KPAD, mt * 16, ks, lane);
      v16bf bb;
      if constexpr (TDMP)
        bb = load_b_ldsf(BL, 64, ks, wave * 16, lane);
      else
        bb = load_b_glb(Bglb + wave * 16, SP, ks, C_in - 1, j, false, lane);
      acc = wmma_bf16(a, bb, acc);
    }
#pragma unroll
    for (int r = 0; r < 8; ++r) {
      int row = mt * 16 + r + ((lane & 16) ? 8 : 0);
      if (row < C_out) {
        long o = b * C_out * SP + (long)row * SP + col0 + wave * 16 + j;
        float v = acc[r] + bias[row];
        if (resid) v += resid[o];
        out[o] = v;
      }
    }
  }
}

// ---------- stage F1: rows x 64 (real) -> rows x 16 (8 complex W-modes) ----------
__global__ void dft_rows_k(const float* __restrict__ in, const float* __restrict__ Bm,
                           float* __restrict__ out, long nrows) {
  __shared__ float BL[64 * 16];
  if (threadIdx.x < 32) {
    tdm_load_2d(BL, Bm, 16, 64, 16);
    __builtin_amdgcn_s_wait_tensorcnt(0);
  }
  __syncthreads();
  long task = (long)blockIdx.x * (blockDim.x >> 5) + (threadIdx.x >> 5);
  int lane = threadIdx.x & 31, j = lane & 15;
  if (task * 16 >= nrows) return;
  const float* arow = in + (task * 16 + (lane & 15)) * 64;
  v8f acc = {};
#pragma unroll
  for (int ks = 0; ks < 64; ks += 32) {
    v16bf a = load_a_row(arow, ks, 64, lane);
    v16bf b = load_b_ldsf(BL, 16, ks, 0, lane);
    acc = wmma_bf16(a, b, acc);
  }
#pragma unroll
  for (int r = 0; r < 8; ++r) {
    long row = task * 16 + r + ((lane & 16) ? 8 : 0);
    out[row * 16 + j] = acc[r];
  }
}

// ---------- batched complex transform: Out = Are*B + Aim*swap(B), per small GEMM --
template <int KPAD, int MROWS>
__global__ void fstage_k(const float* __restrict__ Are, const float* __restrict__ Aim,
                         int kmax,
                         const float* __restrict__ Bbase, int inner, long inner_str,
                         long outer_str, long b_row_str,
                         float* __restrict__ Obase, long o_inner_str, long o_outer_str,
                         long o_row_str, long ntasks) {
  __shared__ float ARL[MROWS * KPAD];
  __shared__ float AIL[MROWS * KPAD];
  if (threadIdx.x < 32) {
    tdm_load_2d(ARL, Are, KPAD, MROWS, KPAD);
    tdm_load_2d(AIL, Aim, KPAD, MROWS, KPAD);
    __builtin_amdgcn_s_wait_tensorcnt(0);
  }
  __syncthreads();
  long task = (long)blockIdx.x * (blockDim.x >> 5) + (threadIdx.x >> 5);
  int lane = threadIdx.x & 31, j = lane & 15;
  if (task >= ntasks) return;
  constexpr int mtiles = MROWS >> 4;
  long g  = task / mtiles;
  int  mt = (int)(task - (long)mtiles * g);
  const float* Bp = Bbase + (g / inner) * outer_str + (g % inner) * inner_str;
  float*       Op = Obase + (g / inner) * o_outer_str + (g % inner) * o_inner_str;
  v8f acc = {};
#pragma unroll
  for (int ks = 0; ks < KPAD; ks += 32) {
    v16bf ar = load_a_ldsf(ARL, KPAD, mt * 16, ks, lane);
    v16bf b0 = load_b_glb(Bp, b_row_str, ks, kmax, j, false, lane);
    acc = wmma_bf16(ar, b0, acc);
    v16bf ai = load_a_ldsf(AIL, KPAD, mt * 16, ks, lane);
    v16bf b1 = load_b_glb(Bp, b_row_str, ks, kmax, j, true, lane);
    acc = wmma_bf16(ai, b1, acc);
  }
#pragma unroll
  for (int r = 0; r < 8; ++r) {
    int row = mt * 16 + r + ((lane & 16) ? 8 : 0);
    Op[(long)row * o_row_str + j] = acc[r];
  }
}

// ---------- per-mode complex channel mixing (tiny flops -> VALU) ----------
__global__ void spectral_k(const float* __restrict__ G3, const float* __restrict__ fw,
                           float* __restrict__ S) {
  int idx = blockIdx.x * blockDim.x + threadIdx.x;   // 2*64*16*16*8 = 262144
  int k  = idx & 7;
  int ky = (idx >> 3) & 15;
  int kz = (idx >> 7) & 15;
  int o  = (idx >> 11) & 63;
  int b  = (idx >> 17) & 1;
  long mo = (long)kz * 256 + ky * 16 + k * 2;
  const float* g = G3 + (long)b * 64 * 4096 + mo;    // stride over i: 4096
  const float* w = fw + (long)o * 4096 + mo;         // stride over i: 262144
  float sr = 0.f, si = 0.f;
  for (int i = 0; i < 64; ++i) {
    if ((i & 15) == 0)   // speculative prefetch of upcoming mode-weight lines
      __builtin_prefetch((const void*)(w + (long)(i + 16) * 262144), 0, 1);
    float gr = g[(long)i * 4096],   gi = g[(long)i * 4096 + 1];
    float wr = w[(long)i * 262144], wi = w[(long)i * 262144 + 1];
    sr += gr * wr - gi * wi;
    si += gr * wi + gi * wr;
  }
  float* s = S + ((long)b * 64 + o) * 4096 + mo;
  s[0] = sr; s[1] = si;
}

// ---------- inverse W stage: rows x 16 -> += rows x 64 real (fused into h_next) ---
__global__ void idft_accum_k(const float* __restrict__ in, const float* __restrict__ Mw,
                             float* __restrict__ out, long nrows) {
  __shared__ float BL[32 * 64];
  if (threadIdx.x < 32) {
    tdm_load_2d(BL, Mw, 64, 32, 64);
    __builtin_amdgcn_s_wait_tensorcnt(0);
  }
  __syncthreads();
  long task = (long)blockIdx.x * (blockDim.x >> 5) + (threadIdx.x >> 5);
  int lane = threadIdx.x & 31, j = lane & 15;
  if (task * 16 >= nrows) return;
  const float* arow = in + (task * 16 + (lane & 15)) * 16;
  v16bf a = load_a_row(arow, 0, 16, lane);   // K padded 16->32 with zeros
#pragma unroll
  for (int nt = 0; nt < 4; ++nt) {
    v8f acc = {};
    v16bf b = load_b_ldsf(BL, 64, 0, nt * 16, lane);
    acc = wmma_bf16(a, b, acc);
#pragma unroll
    for (int r = 0; r < 8; ++r) {
      long row = task * 16 + r + ((lane & 16) ? 8 : 0);
      out[row * 64 + nt * 16 + j] += acc[r];
    }
  }
}

// ---------- group norm ----------
__global__ void zero_k(float* p, int n) {
  int i = blockIdx.x * blockDim.x + threadIdx.x;
  if (i < n) p[i] = 0.f;
}

__global__ void gn_stats_k(const float* __restrict__ y, float* __restrict__ sum,
                           float* __restrict__ sumsq, int cpg, int G) {
  int bg = blockIdx.y;
  const float* base = y + (long)bg * cpg * SP;   // group channels are contiguous
  long n = (long)cpg * SP;
  float s = 0.f, ss = 0.f;
  for (long i = (long)blockIdx.x * blockDim.x + threadIdx.x; i < n;
       i += (long)gridDim.x * blockDim.x) {
    float v = base[i]; s += v; ss += v * v;
  }
  __shared__ float rs[256], rq[256];
  int t = threadIdx.x;
  rs[t] = s; rq[t] = ss;
  __syncthreads();
  for (int off = 128; off > 0; off >>= 1) {
    if (t < off) { rs[t] += rs[t + off]; rq[t] += rq[t + off]; }
    __syncthreads();
  }
  if (t == 0) { atomicAdd(&sum[bg], rs[0]); atomicAdd(&sumsq[bg], rq[0]); }
}

__global__ void gn_final_k(const float* sum, const float* sumsq, float* mu, float* rsig,
                           long n, int count) {
  int i = threadIdx.x;
  if (i < count) {
    float m = sum[i] / (float)n;
    mu[i]   = m;
    rsig[i] = rsqrtf(sumsq[i] / (float)n - m * m + 1e-5f);
  }
}

__global__ void gn_apply_k(float* __restrict__ y, const float* mu, const float* rsig,
                           const float* __restrict__ gamma, const float* __restrict__ beta,
                           int C, int G, long total) {
  long idx = (long)blockIdx.x * blockDim.x + threadIdx.x;
  if (idx >= total) return;
  int c  = (int)((idx / SP) % C);
  int b  = (int)(idx / (SP * C));
  int bg = b * G + c / (C / G);
  float v = (y[idx] - mu[bg]) * rsig[bg] * gamma[c] + beta[c];
  y[idx] = 0.5f * v * (1.f + erff(v * 0.70710678f));   // exact gelu
}

extern "C" void kernel_launch(void* const* d_in, const int* in_sizes, int n_in,
                              void* d_out, int out_size, void* d_ws, size_t ws_size,
                              hipStream_t stream) {
  const float* x     = (const float*)d_in[0];
  const float* in_w  = (const float*)d_in[1];
  const float* in_b  = (const float*)d_in[2];
  const float* in_g  = (const float*)d_in[3];
  const float* in_be = (const float*)d_in[4];
  const float* fw    = (const float*)d_in[5];
  const float* lw    = (const float*)d_in[6];
  const float* lb    = (const float*)d_in[7];
  const float* ow1   = (const float*)d_in[8];
  const float* ob1   = (const float*)d_in[9];
  const float* og    = (const float*)d_in[10];
  const float* obe   = (const float*)d_in[11];
  const float* ow2   = (const float*)d_in[12];
  const float* ob2   = (const float*)d_in[13];
  float* out = (float*)d_out;
  float* W   = (float*)d_ws;

  build_mats_k<<<8, 256, 0, stream>>>(W);

  float* hc = W + OFF_H0;
  float* hn = W + OFF_HN;

  // lifting conv (8->64, K zero-padded to 32) + group norm(8) + gelu
  conv_wmma_k<32, false><<<8192, 128, 0, stream>>>(in_w, in_b, x, nullptr, hc, 64, 8);
  zero_k<<<1, 64, 0, stream>>>(W + OFF_SUM, 64);
  gn_stats_k<<<dim3(64, 16), 256, 0, stream>>>(hc, W + OFF_SUM, W + OFF_SUMSQ, 8, 8);
  gn_final_k<<<1, 32, 0, stream>>>(W + OFF_SUM, W + OFF_SUMSQ, W + OFF_MU, W + OFF_RSIG,
                                   8 * SP, 16);
  gn_apply_k<<<(unsigned)((2L * 64 * SP + 255) / 256), 256, 0, stream>>>(
      hc, W + OFF_MU, W + OFF_RSIG, in_g, in_be, 64, 8, 2L * 64 * SP);

  const long NROWS = 2L * 64 * 64 * 64;   // (b,c,d,h) rows
  for (int l = 0; l < 4; ++l) {
    // forward truncated DFT: W axis (real->8 complex modes)
    dft_rows_k<<<(unsigned)(NROWS / 64), 128, 0, stream>>>(hc, W + OFF_BW, W + OFF_G1,
                                                           NROWS);
    // H axis: per (b,c,d): [16 x 64] x [64 x 16]
    fstage_k<64, 16><<<2048, 128, 0, stream>>>(W + OFF_TREH, W + OFF_TIMH, 63,
                                               W + OFF_G1, 64, 1024, 65536, 16,
                                               W + OFF_G2, 256, 16384, 16, 8192);
    // D axis: per (b,c,ky): [16 x 64] x [64(d,stride 256) x 16]
    fstage_k<64, 16><<<512, 128, 0, stream>>>(W + OFF_TRED, W + OFF_TIMD, 63,
                                              W + OFF_G2, 16, 16, 16384, 256,
                                              W + OFF_G3, 16, 4096, 256, 2048);
    // per-mode complex channel mix
    spectral_k<<<1024, 256, 0, stream>>>(W + OFF_G3, fw + (long)l * 16777216, W + OFF_S);
    // inverse D: per (b,o,ky): [64 x 32(pad)] x [32(kz clamp) x 16]
    fstage_k<32, 64><<<2048, 128, 0, stream>>>(W + OFF_EDRE, W + OFF_EDIM, 15,
                                               W + OFF_S, 16, 16, 4096, 256,
                                               W + OFF_HH1, 16, 16384, 256, 8192);
    // inverse H: per (b,o,d): [64 x 32(pad)] x [32(ky clamp) x 16]
    fstage_k<32, 64><<<8192, 128, 0, stream>>>(W + OFF_EHRE, W + OFF_EHIM, 15,
                                               W + OFF_HH1, 64, 256, 16384, 16,
                                               W + OFF_HH2, 1024, 65536, 16, 32768);
    // skip conv (64->64) + bias + residual, overwrite h_next  (TDM panels)
    conv_wmma_k<64, true><<<8192, 128, 0, stream>>>(lw + (long)l * 4096,
                                                    lb + (long)l * 64, hc, hc, hn,
                                                    64, 64);
    // inverse W (Hermitian irDFT), accumulate into h_next
    idft_accum_k<<<(unsigned)(NROWS / 64), 128, 0, stream>>>(W + OFF_HH2, W + OFF_MW,
                                                             hn, NROWS);
    float* t = hc; hc = hn; hn = t;
  }

  // projection: conv(64->32) + group norm(4) + gelu + conv(32->4)
  float* t1 = hn;   // free ping-pong buffer
  conv_wmma_k<64, true><<<8192, 128, 0, stream>>>(ow1, ob1, hc, nullptr, t1, 32, 64);
  zero_k<<<1, 64, 0, stream>>>(W + OFF_SUM, 64);
  gn_stats_k<<<dim3(64, 8), 256, 0, stream>>>(t1, W + OFF_SUM, W + OFF_SUMSQ, 8, 4);
  gn_final_k<<<1, 32, 0, stream>>>(W + OFF_SUM, W + OFF_SUMSQ, W + OFF_MU, W + OFF_RSIG,
                                   8 * SP, 8);
  gn_apply_k<<<(unsigned)((2L * 32 * SP + 255) / 256), 256, 0, stream>>>(
      t1, W + OFF_MU, W + OFF_RSIG, og, obe, 32, 4, 2L * 32 * SP);
  conv_wmma_k<32, true><<<8192, 128, 0, stream>>>(ow2, ob2, t1, nullptr, out, 4, 32);
}
